// MiniGenerator_13099650253582
// MI455X (gfx1250) — matrix operand order repaired and checked by reading, not compiled, not run on hardware
//
#include <hip/hip_runtime.h>

// CDNA5 / gfx1250 fused conv-autoencoder, LDS-free formulation.
// Each net stage is a dense 32x32 linear map per sample:
//   D[16 feat x 16 samples] = W[16x32] (A) * act[32x16] (B) + bias (C)
// via v_wmma_f32_16x16x32_f16, two WMMAs per stage (feature halves).
// Inter-stage transpose: v_cvt_pk_rtz_f16_f32 + shfl_xor(16) + cndmask.

typedef __attribute__((ext_vector_type(16))) _Float16 v16h;
typedef __attribute__((ext_vector_type(8)))  _Float16 v8h;
typedef __attribute__((ext_vector_type(2)))  __fp16   v2fp16;   // cvt_pkrtz result type
typedef __attribute__((ext_vector_type(8)))  float    v8f;

#define LEAKY_SLOPE 0.2f

union MV { v16h v; v8h p[2]; int i[8]; };
union CV { v8f  v; float f[8]; };
union PK { v2fp16 h2; int i; };

__device__ inline int pk16(float a, float b) {   // one v_cvt_pk_rtz_f16_f32
  PK u; u.h2 = __builtin_amdgcn_cvt_pkrtz(a, b); return u.i;
}

// ---------------------------------------------------------------------------
// Weight densification: conv (+stride/+upsample) folded to a 32x32 matrix per
// stage, stored in WMMA *A-register image* order:
//   WA[stage][half][lane][j],  j = 0..15
//   element = W[n = (lane&15) + 16*half][k = j + 16*(lane>>4)]
// ---------------------------------------------------------------------------
__device__ float w_entry(int s, int k, int n,
                         const float* w1, const float* w2,
                         const float* w3, const float* w4) {
  float acc = 0.f;
  if (s == 0) {            // enc1: x[2][16] -> [4][8], stride2 pad1
    int co = n >> 3, p = n & 7;
    int ci = k >> 4, i = k & 15;
    for (int t = 0; t < 3; ++t)
      if (2 * p - 1 + t == i) acc += w1[(co * 2 + ci) * 3 + t];
  } else if (s == 1) {     // bneck: [4][8] -> [8][4], stride2 pad1
    int co = n >> 2, p = n & 3;
    int ci = k >> 3, i = k & 7;
    for (int t = 0; t < 3; ++t)
      if (2 * p - 1 + t == i) acc += w2[(co * 4 + ci) * 3 + t];
  } else if (s == 2) {     // dec1: upsample2(bneck [8][4]) -> [4][8], stride1 pad1
    int co = n >> 3, p = n & 7;
    int ci = k >> 2, q = k & 3;
    for (int t = 0; t < 3; ++t) {
      int j = p - 1 + t;
      if (j >= 0 && j < 8 && (j >> 1) == q) acc += w3[(co * 8 + ci) * 3 + t];
    }
  } else {                 // out: upsample2(skip [4][8]) -> [2][16], stride1 pad1
    int co = n >> 4, i = n & 15;
    int ci = k >> 3, q = k & 7;
    for (int t = 0; t < 3; ++t) {
      int j = i - 1 + t;
      if (j >= 0 && j < 16 && (j >> 1) == q) acc += w4[(co * 4 + ci) * 3 + t];
    }
  }
  return acc;
}

__global__ void build_weights(const float* __restrict__ w1, const float* __restrict__ b1,
                              const float* __restrict__ w2, const float* __restrict__ b2,
                              const float* __restrict__ w3, const float* __restrict__ b3,
                              const float* __restrict__ w4, const float* __restrict__ b4,
                              _Float16* __restrict__ WA, float* __restrict__ bias) {
  int idx = blockIdx.x * blockDim.x + threadIdx.x;
  if (idx < 4096) {
    int s   = idx >> 10;
    int rem = idx & 1023;
    int h   = rem >> 9;
    int l   = (rem >> 4) & 31;
    int j   = rem & 15;
    int k   = j + ((l >> 4) << 4);                     // A-layout K row
    int n   = (l & 15) + (h << 4);                     // output feature (A-row M)
    WA[idx] = (_Float16)w_entry(s, k, n, w1, w2, w3, w4);
  } else if (idx < 4224) {
    int b = idx - 4096;
    int s = b >> 5, n = b & 31;
    float v;
    if      (s == 0) v = b1[n >> 3];
    else if (s == 1) v = b2[n >> 2];
    else if (s == 2) v = b3[n >> 3];
    else             v = b4[n >> 4];
    bias[b] = v;
  }
}

// ---------------------------------------------------------------------------
__device__ inline float leaky(float u) {
  return u > 0.f ? u : LEAKY_SLOPE * u;   // v_cmp + v_cndmask/v_mul, no canonicalize
}

__device__ inline float fast_tanh(float x) {
  x = fminf(fmaxf(x, -10.f), 10.f);
  float e = __expf(2.f * x);
  return (e - 1.f) / (e + 1.f);
}

// D[feat][sample] (f32, two halves) -> next stage's B operand [K feat][N sample].
// Lane l and lane l^16 hold the same sample; swap packed halves across them.
__device__ inline MV transition(const CV& alo, const CV& ahi, bool lower) {
  int plo[4], phi[4];
#pragma unroll
  for (int q = 0; q < 4; ++q) {
    plo[q] = pk16(alo.f[2 * q], alo.f[2 * q + 1]);
    phi[q] = pk16(ahi.f[2 * q], ahi.f[2 * q + 1]);
  }
  MV b;
#pragma unroll
  for (int q = 0; q < 4; ++q) {
    int swlo = __shfl_xor(plo[q], 16, 32);
    int swhi = __shfl_xor(phi[q], 16, 32);
    b.i[q]     = lower ? plo[q] : swhi;       // K 0..7   | K 16..23
    b.i[4 + q] = lower ? swlo   : phi[q];     // K 8..15  | K 24..31
  }
  return b;
}

#define WMMA(Aop, Bop, Cop) \
  __builtin_amdgcn_wmma_f32_16x16x32_f16(false, (Aop), false, (Bop), (short)0, (Cop), false, false)

__global__ void __launch_bounds__(256)
unet_fused(const float* __restrict__ x, const _Float16* __restrict__ WA,
           const float* __restrict__ bias, float* __restrict__ out, int ntiles) {
  const int  lane  = threadIdx.x & 31;
  const bool lower = lane < 16;

  // wave-uniform loop bounds in SGPRs -> scalar branch, EXEC stays all-ones
  const int nwaves = (gridDim.x * blockDim.x) >> 5;
  const int gwave  = __builtin_amdgcn_readfirstlane(
                        (int)((blockIdx.x * blockDim.x + threadIdx.x) >> 5));

  // ---- resident weights (A operands): 4 stages x 2 halves, b128 loads
  MV A[4][2];
#pragma unroll
  for (int s = 0; s < 4; ++s)
#pragma unroll
    for (int h = 0; h < 2; ++h) {
      const v8h* p = (const v8h*)(WA + (((s * 2 + h) * 32 + lane) << 4));
      A[s][h].p[0] = p[0];
      A[s][h].p[1] = p[1];
    }

  // ---- resident biases as WMMA C operands: bias varies along M (= register r)
  CV cb[4][2];
#pragma unroll
  for (int s = 0; s < 4; ++s) {
    const float* bp = bias + s * 32 + ((lane >> 4) << 3);
#pragma unroll
    for (int h = 0; h < 2; ++h) {
      float4 f0 = *(const float4*)(bp + 16 * h);
      float4 f1 = *(const float4*)(bp + 16 * h + 4);
      cb[s][h].f[0] = f0.x; cb[s][h].f[1] = f0.y; cb[s][h].f[2] = f0.z; cb[s][h].f[3] = f0.w;
      cb[s][h].f[4] = f1.x; cb[s][h].f[5] = f1.y; cb[s][h].f[6] = f1.z; cb[s][h].f[7] = f1.w;
    }
  }

  // per-lane input offset: sample (lane&15), 16 consecutive feats at 16*(lane>>4)
  const int soff = ((lane & 15) << 5) + ((lane >> 4) << 4);
  const int fo   = (lane >> 4) << 3;          // output feature offset within halves

  for (int tile = gwave; tile < ntiles; tile += nwaves) {
    const float* xt = x + ((size_t)tile << 9) + soff;
    if (tile + nwaves < ntiles)               // uniform: global_prefetch_b8
      __builtin_prefetch(x + ((size_t)(tile + nwaves) << 9) + soff, 0, 0);

    // ---- global -> B operand (4 x b128 per lane, no LDS)
    float4 q0 = *(const float4*)(xt);
    float4 q1 = *(const float4*)(xt + 4);
    float4 q2 = *(const float4*)(xt + 8);
    float4 q3 = *(const float4*)(xt + 12);
    MV b;
    b.i[0] = pk16(q0.x, q0.y); b.i[1] = pk16(q0.z, q0.w);
    b.i[2] = pk16(q1.x, q1.y); b.i[3] = pk16(q1.z, q1.w);
    b.i[4] = pk16(q2.x, q2.y); b.i[5] = pk16(q2.z, q2.w);
    b.i[6] = pk16(q3.x, q3.y); b.i[7] = pk16(q3.z, q3.w);

    CV lo, hi, a1lo, a1hi, alo, ahi;

    // ---- stage 1: enc1 = leaky(W1 x + b1)       (kept in regs for the skip)
    lo.v = WMMA(A[0][0].v, b.v, cb[0][0].v);
    hi.v = WMMA(A[0][1].v, b.v, cb[0][1].v);
#pragma unroll
    for (int r = 0; r < 8; ++r) {
      a1lo.f[r] = leaky(lo.f[r]);
      a1hi.f[r] = leaky(hi.f[r]);
    }
    b = transition(a1lo, a1hi, lower);

    // ---- stage 2: bneck = leaky(W2 enc1 + b2)
    lo.v = WMMA(A[1][0].v, b.v, cb[1][0].v);
    hi.v = WMMA(A[1][1].v, b.v, cb[1][1].v);
#pragma unroll
    for (int r = 0; r < 8; ++r) {
      alo.f[r] = leaky(lo.f[r]);
      ahi.f[r] = leaky(hi.f[r]);
    }
    b = transition(alo, ahi, lower);

    // ---- stage 3: skip = leaky(W3 up(bneck) + b3) + enc1   (upsample in W3)
    lo.v = WMMA(A[2][0].v, b.v, cb[2][0].v);
    hi.v = WMMA(A[2][1].v, b.v, cb[2][1].v);
#pragma unroll
    for (int r = 0; r < 8; ++r) {
      alo.f[r] = leaky(lo.f[r]) + a1lo.f[r];
      ahi.f[r] = leaky(hi.f[r]) + a1hi.f[r];
    }
    b = transition(alo, ahi, lower);

    // ---- stage 4: out = tanh(W4 up(skip) + b4)   (upsample in W4)
    lo.v = WMMA(A[3][0].v, b.v, cb[3][0].v);
    hi.v = WMMA(A[3][1].v, b.v, cb[3][1].v);
#pragma unroll
    for (int r = 0; r < 8; ++r) {
      alo.f[r] = fast_tanh(lo.f[r]);
      ahi.f[r] = fast_tanh(hi.f[r]);
    }

    // ---- registers -> global (4 x b128 per lane, no LDS)
    float* ot = out + ((size_t)tile << 9) + ((lane & 15) << 5);
    *(float4*)(ot + fo)      = make_float4(alo.f[0], alo.f[1], alo.f[2], alo.f[3]);
    *(float4*)(ot + fo + 4)  = make_float4(alo.f[4], alo.f[5], alo.f[6], alo.f[7]);
    *(float4*)(ot + fo + 16) = make_float4(ahi.f[0], ahi.f[1], ahi.f[2], ahi.f[3]);
    *(float4*)(ot + fo + 20) = make_float4(ahi.f[4], ahi.f[5], ahi.f[6], ahi.f[7]);
  }
}

// ---------------------------------------------------------------------------
extern "C" void kernel_launch(void* const* d_in, const int* in_sizes, int n_in,
                              void* d_out, int out_size, void* d_ws, size_t ws_size,
                              hipStream_t stream) {
  const float* x  = (const float*)d_in[0];
  const float* w1 = (const float*)d_in[1];
  const float* b1 = (const float*)d_in[2];
  const float* w2 = (const float*)d_in[3];
  const float* b2 = (const float*)d_in[4];
  const float* w3 = (const float*)d_in[5];
  const float* b3 = (const float*)d_in[6];
  const float* w4 = (const float*)d_in[7];
  const float* b4 = (const float*)d_in[8];
  float* out = (float*)d_out;

  _Float16* WA = (_Float16*)d_ws;                   // 4096 f16 = 8 KB
  float* bias  = (float*)((char*)d_ws + 8192);      // 128 f32

  int nsamples = in_sizes[0] / 32;                  // B = 262144
  int ntiles   = nsamples / 16;                     // 16384 tiles of 16 samples

  build_weights<<<17, 256, 0, stream>>>(w1, b1, w2, b2, w3, b3, w4, b4, WA, bias);

  int blocks = 2048;                                // 8 waves/block -> 16384 waves
  unet_fused<<<blocks, 256, 0, stream>>>(x, WA, bias, out, ntiles);
}
